// LinearAttention_8100308321079
// MI455X (gfx1250) — matrix operand order repaired and checked by reading, not compiled, not run on hardware
//
#include <hip/hip_runtime.h>
#include <hip/hip_bf16.h>

// Problem constants (B,H,L,D) = (2,16,2048,64)
#define B_   2
#define H_   16
#define L_   2048
#define D_   64
#define BH_  (B_*H_)
#define NSTRIP (L_/16)   // 128 strips of 16 rows

typedef __attribute__((ext_vector_type(16))) __bf16 v16bf;
typedef __attribute__((ext_vector_type(8)))  __bf16 v8bf;
typedef __attribute__((ext_vector_type(8)))  float  v8f;

// Native conversion: lowers to v_cvt_(pk_)bf16_f32 on gfx1250-class targets.
static __device__ __forceinline__ unsigned short f2bfu(float f) {
    __bf16 b = (__bf16)f;
    return __builtin_bit_cast(unsigned short, b);
}

static __device__ __forceinline__ v16bf cat8(v8bf lo, v8bf hi) {
    return __builtin_shufflevector(lo, hi, 0,1,2,3,4,5,6,7,8,9,10,11,12,13,14,15);
}

// ---------------------------------------------------------------------------
// Fragment loaders (CDNA5 wave32 WMMA layouts, CDNA5 ISA §7.12.2)
//
// 16-bit A 16x32 (MxK): lane m=lane&15; lanes<16 hold K {0..7,16..23},
//                       lanes>=16 hold K {8..15,24..31}  -> two b128 loads.
// 16-bit B 32x16 (KxN): lane n=lane&15; lanes<16 hold K {0..7,16..23},
//                       lanes>=16 hold K {8..15,24..31}  -> two b128 loads
//                       when B = M^T of a row-major M (contiguous in a row).
// fp32  C/D 16x16:      VGPR v: row = v + (lane<16?0:8), col = lane&15.
// ---------------------------------------------------------------------------

// A fragment from row-major bf16, 16B-aligned rows; base at (row0, k0)
static __device__ __forceinline__ v16bf load_a_bf16v(const unsigned short* base, int ld, int lane) {
    const int m    = lane & 15;
    const int koff = (lane < 16) ? 0 : 8;
    const __bf16* row = (const __bf16*)(base) + m * ld + koff;
    v8bf lo = *(const v8bf*)(row);
    v8bf hi = *(const v8bf*)(row + 16);
    return cat8(lo, hi);
}

// B(k,n) = M[n][k] fragment for row-major bf16 M (B = M^T); base at (n0, k0)
static __device__ __forceinline__ v16bf load_bT_bf16v(const unsigned short* base, int ld, int lane) {
    const int n   = lane & 15;
    const int kk0 = (lane < 16) ? 0 : 8;
    const __bf16* row = (const __bf16*)(base) + n * ld + kk0;
    v8bf lo = *(const v8bf*)(row);
    v8bf hi = *(const v8bf*)(row + 16);
    return cat8(lo, hi);
}

// ---------------------------------------------------------------------------
// Kernel 0: streaming fp32 -> bf16 convert (float4 in, 2 packed dwords out).
// grid-stride free: exactly N/4 threads. Used for Q and K.
// ---------------------------------------------------------------------------
__global__ __launch_bounds__(256)
void la_cvt_kernel(const float* __restrict__ X, unsigned short* __restrict__ Xb) {
    const size_t i4 = (((size_t)blockIdx.x * 256) + threadIdx.x) * 4;
    const float4 v = *(const float4*)(X + i4);
    unsigned r0 = (unsigned)f2bfu(v.x) | ((unsigned)f2bfu(v.y) << 16);
    unsigned r1 = (unsigned)f2bfu(v.z) | ((unsigned)f2bfu(v.w) << 16);
    uint2 o; o.x = r0; o.y = r1;
    *(uint2*)(Xb + i4) = o;
}

// ---------------------------------------------------------------------------
// Kernel 1: per-head weight transpose+convert: Wt[h][n][k] = W[h][k][n] (bf16)
// so featmap B fragments become contiguous b128 loads. 32 blocks (2 W's x 16 h).
// ---------------------------------------------------------------------------
__global__ __launch_bounds__(256)
void la_wt_kernel(const float* __restrict__ Wq, const float* __restrict__ Wk,
                  unsigned short* __restrict__ Wqt, unsigned short* __restrict__ Wkt) {
    __shared__ float tile[64][65];             // +1 pad: no bank conflicts
    const int which = blockIdx.x >> 4;
    const int h     = blockIdx.x & 15;
    const float* src = (which ? Wk : Wq) + (size_t)h * D_ * D_;
    unsigned short* dst = (which ? Wkt : Wqt) + (size_t)h * D_ * D_;
    for (int i = threadIdx.x; i < 4096; i += 256) {
        const int d = i >> 6, e = i & 63;
        tile[e][d] = src[i];                   // coalesced read, transposed store
    }
    __syncthreads();
    for (int i = threadIdx.x; i < 4096; i += 256)
        dst[i] = f2bfu(tile[i >> 6][i & 63]);  // coalesced write
}

// ---------------------------------------------------------------------------
// Kernel 2: transpose+convert V -> Vt[bh][d][l] (bf16) via LDS tiles so the
// PV GEMM's B fragments become contiguous b128 loads.
// ---------------------------------------------------------------------------
__global__ __launch_bounds__(256)
void la_vt_kernel(const float* __restrict__ V, unsigned short* __restrict__ Vt) {
    __shared__ unsigned short tile[64][65];    // +1 pad: no bank conflicts
    const int bh = blockIdx.x >> 5;
    const int lt = blockIdx.x & 31;
    const float* src = V + ((size_t)bh * L_ + lt * 64) * D_;
    for (int i = threadIdx.x; i < 4096; i += 256) {
        const int l = i >> 6, d = i & 63;
        tile[d][l] = f2bfu(src[(size_t)l * D_ + d]);   // coalesced read over d
    }
    __syncthreads();
    unsigned short* dst = Vt + (size_t)bh * D_ * L_ + lt * 64;
    for (int i = threadIdx.x; i < 4096; i += 256) {
        const int d = i >> 6, l = i & 63;
        dst[(size_t)d * L_ + l] = tile[d][l];          // coalesced write over l
    }
}

// ---------------------------------------------------------------------------
// Kernel 3: feature maps, all-bf16.  fq = relu(Qb @ Wkt^T), fk = relu(Kb @ Wqt^T)
// (swap faithful to the reference).  One wave per 16x64 output strip; every
// fragment is two b128 loads.
// ---------------------------------------------------------------------------
__global__ __launch_bounds__(32)
void la_featmap_kernel(const unsigned short* __restrict__ Qb,
                       const unsigned short* __restrict__ Kb,
                       const unsigned short* __restrict__ Wqt,
                       const unsigned short* __restrict__ Wkt,
                       unsigned short* __restrict__ fq, unsigned short* __restrict__ fk) {
    const int bid   = blockIdx.x;
    const int which = bid >> 12;            // 0: fq, 1: fk
    const int rem   = bid & 4095;
    const int bh    = rem >> 7;
    const int s     = rem & 127;
    const int h     = bh & (H_ - 1);
    const int lane  = threadIdx.x;

    const unsigned short* X  = (which == 0) ? Qb  : Kb;
    const unsigned short* Wt = (which == 0) ? Wkt : Wqt;   // swapped, per reference
    unsigned short* dst = (which == 0) ? fq : fk;

    const unsigned short* Xs = X + ((size_t)bh * L_ + s * 16) * D_;
    const unsigned short* Wh = Wt + (size_t)h * D_ * D_;   // Wt[h][n][k], ld = 64

    const v16bf a0 = load_a_bf16v(Xs +  0, D_, lane);
    const v16bf a1 = load_a_bf16v(Xs + 32, D_, lane);

    const int n      = lane & 15;
    const int rowoff = (lane < 16) ? 0 : 8;

#pragma unroll
    for (int nt = 0; nt < 4; ++nt) {
        v8f acc = {};
        v16bf b0 = load_bT_bf16v(Wh + nt * 16 * D_ +  0, D_, lane);
        acc = __builtin_amdgcn_wmma_f32_16x16x32_bf16(false, a0, false, b0,
                                                      (short)0, acc, false, false);
        v16bf b1 = load_bT_bf16v(Wh + nt * 16 * D_ + 32, D_, lane);
        acc = __builtin_amdgcn_wmma_f32_16x16x32_bf16(false, a1, false, b1,
                                                      (short)0, acc, false, false);
        unsigned short* drow = dst + ((size_t)bh * L_ + s * 16) * D_ + nt * 16 + n;
#pragma unroll
        for (int v = 0; v < 8; ++v)
            drow[(size_t)(v + rowoff) * D_] = f2bfu(fmaxf(acc[v], 0.0f));
    }
}

// ---------------------------------------------------------------------------
// Kernel 4: fused attention.  One block (4 waves) owns a 16-row strip:
//   LDS fp32 strip = tril(fq_strip @ fk^T) -> LDS row sums -> normalize ->
//   one fp32 attn write to HBM + bf16 copy in LDS -> strip @ V via WMMA.
// LDS: 128 KB fp32 strip + 64 KB bf16 strip (CDNA5 WGP has 320 KB).
// ---------------------------------------------------------------------------
__global__ __launch_bounds__(128)
void la_attn_kernel(const unsigned short* __restrict__ Vt,
                    const unsigned short* __restrict__ fq,
                    const unsigned short* __restrict__ fk,
                    float* __restrict__ out, float* __restrict__ attn) {
    __shared__ float          attnF[16 * L_];   // 128 KB
    __shared__ unsigned short attnB[16 * L_];   //  64 KB
    __shared__ float rowpart[16][8];
    __shared__ float inv16[16];

    const int bh   = blockIdx.x >> 7;
    const int s    = blockIdx.x & 127;
    const int wave = threadIdx.x >> 5;
    const int lane = threadIdx.x & 31;
    const int rmax = s * 16 + 15;

    // ---- Phase A: attn strip tiles (causal skip above the diagonal) ----
    const unsigned short* fqs = fq + ((size_t)bh * L_ + s * 16) * D_;
    const v16bf aq0 = load_a_bf16v(fqs +  0, D_, lane);
    const v16bf aq1 = load_a_bf16v(fqs + 32, D_, lane);

    const int n      = lane & 15;
    const int rowoff = (lane < 16) ? 0 : 8;

    for (int ct = 0; ct < 32; ++ct) {
        const int col0 = wave * 512 + ct * 16;
        const int c    = col0 + n;
        if (col0 > rmax) {                   // fully masked tile: zero-fill LDS
#pragma unroll
            for (int v = 0; v < 8; ++v)
                attnF[(v + rowoff) * L_ + c] = 0.0f;
            continue;
        }
        const unsigned short* fks = fk + ((size_t)bh * L_ + col0) * D_;
        __builtin_prefetch(fks + 16 * D_, 0, 1);   // next fk tile

        v8f acc = {};
        v16bf b0 = load_bT_bf16v(fks +  0, D_, lane);
        acc = __builtin_amdgcn_wmma_f32_16x16x32_bf16(false, aq0, false, b0,
                                                      (short)0, acc, false, false);
        v16bf b1 = load_bT_bf16v(fks + 32, D_, lane);
        acc = __builtin_amdgcn_wmma_f32_16x16x32_bf16(false, aq1, false, b1,
                                                      (short)0, acc, false, false);
#pragma unroll
        for (int v = 0; v < 8; ++v) {
            const int m = v + rowoff;
            attnF[m * L_ + c] = (c <= s * 16 + m) ? acc[v] : 0.0f;
        }
    }
    __syncthreads();

    // ---- Phase B: row sums from the LDS strip (float4 loads), reciprocal ----
    {
        const int row = threadIdx.x >> 3;
        const int seg = threadIdx.x & 7;
        const float* srow = &attnF[row * L_ + seg * 256];
        float sum = 0.0f;
        for (int c = 0; c < 256; c += 4) {
            const float4 q = *(const float4*)&srow[c];
            sum += (q.x + q.y) + (q.z + q.w);
        }
        rowpart[row][seg] = sum;
    }
    __syncthreads();
    if (threadIdx.x < 16) {
        const float* rp = rowpart[threadIdx.x];
        float sum = ((rp[0] + rp[1]) + (rp[2] + rp[3])) +
                    ((rp[4] + rp[5]) + (rp[6] + rp[7]));
        inv16[threadIdx.x] = 1.0f / sum;     // faithful: ref divides directly
    }
    __syncthreads();

    // ---- Phase C: normalize -> fp32 attn to HBM (b128) + bf16 copy in LDS ----
    {
        const int row = threadIdx.x >> 3;
        const int c0  = (threadIdx.x & 7) * 256;
        const float is = inv16[row];
        float*          dstrow = attn + ((size_t)bh * L_ + s * 16 + row) * L_;
        const float*    srow   = &attnF[row * L_];
        unsigned short* brow   = &attnB[row * L_];
        for (int c = c0; c < c0 + 256; c += 4) {
            const float v0 = srow[c + 0] * is;
            const float v1 = srow[c + 1] * is;
            const float v2 = srow[c + 2] * is;
            const float v3 = srow[c + 3] * is;
            float4 o; o.x = v0; o.y = v1; o.z = v2; o.w = v3;
            *(float4*)&dstrow[c] = o;
            *(unsigned*)&brow[c + 0] = (unsigned)f2bfu(v0) | ((unsigned)f2bfu(v1) << 16);
            *(unsigned*)&brow[c + 2] = (unsigned)f2bfu(v2) | ((unsigned)f2bfu(v3) << 16);
        }
    }
    __syncthreads();

    // ---- Phase D: out strip = attn_strip @ V, pure b128 fragment loads ----
    {
        const unsigned short* Vth = Vt + (size_t)bh * D_ * L_ + (size_t)(wave * 16) * L_;
        const int ktend = rmax >> 5;        // highest K-tile with nonzero attn
        v8f acc = {};
        for (int kt = 0; kt <= ktend; ++kt) {
            v16bf a = load_a_bf16v(attnB + kt * 32, L_, lane);      // LDS b128
            v16bf b = load_bT_bf16v(Vth + kt * 32, L_, lane);       // B = Vt^T
            acc = __builtin_amdgcn_wmma_f32_16x16x32_bf16(false, a, false, b,
                                                          (short)0, acc, false, false);
        }
        float* orow = out + ((size_t)bh * L_ + s * 16) * D_ + wave * 16 + n;
#pragma unroll
        for (int v = 0; v < 8; ++v)
            orow[(size_t)(v + rowoff) * D_] = acc[v];
    }
}

// ---------------------------------------------------------------------------
extern "C" void kernel_launch(void* const* d_in, const int* in_sizes, int n_in,
                              void* d_out, int out_size, void* d_ws, size_t ws_size,
                              hipStream_t stream) {
    const float* Q  = (const float*)d_in[0];   // query_states
    const float* K  = (const float*)d_in[1];   // key_states
    const float* V  = (const float*)d_in[2];   // value_states
    const float* Wq = (const float*)d_in[3];   // fm_q_weight
    const float* Wk = (const float*)d_in[4];   // fm_k_weight

    const size_t NT = (size_t)BH_ * L_ * D_;   // 4,194,304 elements

    // workspace (bf16/ushort): fq, fk, Vt, Qb, Kb (8 MB each) + Wt's (tiny)
    unsigned short* fq  = (unsigned short*)d_ws;
    unsigned short* fk  = fq  + NT;
    unsigned short* Vt  = fk  + NT;
    unsigned short* Qb  = Vt  + NT;
    unsigned short* Kb  = Qb  + NT;
    unsigned short* Wqt = Kb  + NT;
    unsigned short* Wkt = Wqt + (size_t)H_ * D_ * D_;

    float* out  = (float*)d_out;                       // [B,H,L,D] fp32
    float* attn = out + NT;                            // [B,H,L,L] fp32

    la_cvt_kernel<<<NT / 1024, 256, 0, stream>>>(Q, Qb);
    la_cvt_kernel<<<NT / 1024, 256, 0, stream>>>(K, Kb);
    la_wt_kernel<<<2 * H_, 256, 0, stream>>>(Wq, Wk, Wqt, Wkt);
    la_vt_kernel<<<BH_ * (L_ / 64), 256, 0, stream>>>(V, Vt);
    la_featmap_kernel<<<2 * BH_ * NSTRIP, 32, 0, stream>>>(Qb, Kb, Wqt, Wkt, fq, fk);
    la_attn_kernel<<<BH_ * NSTRIP, 128, 0, stream>>>(Vt, fq, fk, out, attn);
}
